// EntAttentionLayer_82815559401987
// MI455X (gfx1250) — matrix operand order repaired and checked by reading, not compiled, not run on hardware
//
#include <hip/hip_runtime.h>
#include <hip/hip_bf16.h>
#include <math.h>

#define DEV __device__ __forceinline__

typedef __attribute__((ext_vector_type(16))) __bf16 v16bf;
typedef __attribute__((ext_vector_type(8)))  float  v8f;

static constexpr int Bsz = 2, Sseq = 2048, Dm = 768, Hh = 12, DHd = 64;
static constexpr int Ttag = 50, Rwin = 50, Dff = 3072;
static constexpr int Mrows = Bsz * Sseq; // 4096

// ---- CDNA5 async global->LDS copy (ASYNCcnt), with safe fallback ----
#if defined(__has_builtin)
#if __has_builtin(__builtin_amdgcn_global_load_async_to_lds_b128)
#define HAVE_ASYNC_CP 1
#endif
#endif
#ifndef HAVE_ASYNC_CP
#define HAVE_ASYNC_CP 0
#endif

#if HAVE_ASYNC_CP
typedef int v4i __attribute__((vector_size(16)));
typedef __attribute__((address_space(1))) v4i* gas_v4i;
typedef __attribute__((address_space(3))) v4i* las_v4i;
#endif

// copy 16 bytes global -> LDS (per-lane addresses)
DEV void cp16(const unsigned short* g, unsigned short* l) {
#if HAVE_ASYNC_CP
  __builtin_amdgcn_global_load_async_to_lds_b128(
      (gas_v4i)(unsigned long long)g, (las_v4i)(unsigned)(unsigned long long)l,
      0, 0);
#else
  *(uint4*)l = *(const uint4*)g;
#endif
}

DEV void async_wait0() {
#if HAVE_ASYNC_CP
#if __has_builtin(__builtin_amdgcn_s_wait_asynccnt)
  __builtin_amdgcn_s_wait_asynccnt(0);
#else
  asm volatile("s_wait_asynccnt 0x0" ::: "memory");
#endif
#endif
}

// ---------------- helpers ----------------

DEV unsigned short f2bf(float f) {
  union { float f; unsigned u; } x; x.f = f;
  unsigned r = x.u + 0x7FFFu + ((x.u >> 16) & 1u); // RNE
  return (unsigned short)(r >> 16);
}

DEV v16bf load_frag(const unsigned short* p0, const unsigned short* p1) {
  union { uint4 u[2]; v16bf v; } f;
  f.u[0] = *(const uint4*)p0;
  f.u[1] = *(const uint4*)p1;
  return f.v;
}

DEV v8f wmma_bf16(v16bf a, v16bf b, v8f c) {
  return __builtin_amdgcn_wmma_f32_16x16x32_bf16(false, a, false, b, (short)0, c,
                                                 false, false);
}

// ---------------- conversion kernels ----------------

__global__ void conv_bf16_k(const float* __restrict__ in,
                            unsigned short* __restrict__ out, long n) {
  long i = (long)blockIdx.x * blockDim.x + threadIdx.x;
  if (i < n) out[i] = f2bf(in[i]);
}

// W is [K,N] row-major f32; Wt is [N,K] row-major bf16 (i.e. W^T)
__global__ void transpose_conv_k(const float* __restrict__ W,
                                 unsigned short* __restrict__ Wt, int K, int N) {
  long i = (long)blockIdx.x * blockDim.x + threadIdx.x;
  if (i < (long)K * N) {
    int k = (int)(i % K), n = (int)(i / K);
    Wt[i] = f2bf(W[(size_t)k * N + n]);
  }
}

__global__ void tags_pad_k(const float* __restrict__ tag,
                           unsigned short* __restrict__ out) {
  int i = blockIdx.x * blockDim.x + threadIdx.x;
  if (i < 64 * Dm) {
    int row = i / Dm;
    out[i] = (row < Ttag) ? f2bf(tag[i]) : (unsigned short)0;
  }
}

// ---------------- generic bf16 WMMA GEMM ----------------
// C[M,N] = A[M,K](bf16 rm) * Bt[N,K]^T (bf16) (+bias)(+resid)(+gelu)
// BM=128, BN=64, BK=64; 256 thr = 8 waves; wave = 16 rows x 64 cols.
// Double-buffered LDS; async copies overlap next tile with current WMMAs.

template <bool GUARD>
DEV void gemm_stage(const unsigned short* __restrict__ A,
                    const unsigned short* __restrict__ Bt,
                    unsigned short* sAb, unsigned short* sBb, int row0, int n0,
                    int M, int K, int kb, int tid) {
  // A tile: 128 rows x 64 halves (128B) = 8 x 16B chunks per row
#pragma unroll
  for (int j = 0; j < 4; ++j) {
    int c = tid + j * 256;
    int r = c >> 3, sl = c & 7;
    const unsigned short* gp = A + (size_t)(row0 + r) * K + kb + sl * 8;
    unsigned short* lp = &sAb[r * 72 + sl * 8];
    if (GUARD) {
      uint4 v = make_uint4(0u, 0u, 0u, 0u);
      if (row0 + r < M) v = *(const uint4*)gp;
      *(uint4*)lp = v;
    } else {
      cp16(gp, lp);
    }
  }
  // B tile: 64 rows(n) x 64 halves = 8 chunks per row
#pragma unroll
  for (int j = 0; j < 2; ++j) {
    int c = tid + j * 256;
    int r = c >> 3, sl = c & 7;
    const unsigned short* gp = Bt + (size_t)(n0 + r) * K + kb + sl * 8;
    unsigned short* lp = &sBb[r * 72 + sl * 8];
    if (GUARD) *(uint4*)lp = *(const uint4*)gp;
    else cp16(gp, lp);
  }
}

template <bool GUARD>
__global__ __launch_bounds__(256) void gemm_bf16_k(
    const unsigned short* __restrict__ A, const unsigned short* __restrict__ Bt,
    const float* __restrict__ bias, const float* __restrict__ resid,
    float* __restrict__ outF, unsigned short* __restrict__ outB,
    int M, int N, int K, int fuse_gelu) {
  __shared__ unsigned short sA[2][128 * 72]; // 64 halves + 8 pad
  __shared__ unsigned short sB[2][64 * 72];

  const int tid = threadIdx.x, lane = tid & 31, wid = tid >> 5;
  const int row0 = blockIdx.x * 128, n0 = blockIdx.y * 64;
  const int mrow = lane & 15, hi = (lane >> 4) & 1;
  const int kA = hi ? 8 : 0;   // A frag: lanes16-31 start at K=8
  const int kB = hi ? 8 : 0;   // B frag: lanes16-31 start at K=8

  v8f z = {0.f, 0.f, 0.f, 0.f, 0.f, 0.f, 0.f, 0.f};
  v8f acc[4] = {z, z, z, z};

  const int nst = K >> 6;
  gemm_stage<GUARD>(A, Bt, sA[0], sB[0], row0, n0, M, K, 0, tid);

  for (int i = 0; i < nst; ++i) {
    const unsigned short* cA = sA[i & 1];
    const unsigned short* cB = sB[i & 1];
    async_wait0();        // my copies for buffer (i&1) have landed
    __syncthreads();      // everyone's copies landed; prev compute finished
    if (i + 1 < nst)
      gemm_stage<GUARD>(A, Bt, sA[(i + 1) & 1], sB[(i + 1) & 1], row0, n0, M, K,
                        (i + 1) << 6, tid);

    int m = wid * 16 + mrow;
#pragma unroll
    for (int ks = 0; ks < 2; ++ks) {
      const unsigned short* ab = &cA[m * 72 + ks * 32 + kA];
      v16bf a = load_frag(ab, ab + 16);
#pragma unroll
      for (int nt = 0; nt < 4; ++nt) {
        const unsigned short* bb = &cB[(nt * 16 + mrow) * 72 + ks * 32 + kB];
        v16bf bf = load_frag(bb, bb + 16);
        acc[nt] = wmma_bf16(a, bf, acc[nt]);
      }
    }
  }

  // epilogue (C layout: VGPR r -> row r + 8*hi, col = lane&15)
#pragma unroll
  for (int nt = 0; nt < 4; ++nt) {
    int gcol = n0 + nt * 16 + mrow;
    float bv = bias ? bias[gcol] : 0.0f;
#pragma unroll
    for (int r = 0; r < 8; ++r) {
      int grow = row0 + wid * 16 + r + hi * 8;
      if (GUARD && grow >= M) continue;
      float v = acc[nt][r] + bv;
      if (resid) v += resid[(size_t)grow * N + gcol];
      if (fuse_gelu) v = 0.5f * v * (1.0f + erff(v * 0.70710678118654752f));
      if (outF) outF[(size_t)grow * N + gcol] = v;
      if (outB) outB[(size_t)grow * N + gcol] = f2bf(v);
    }
  }
}

// ---------------- flash attention ----------------
// grid = (S/128, H, B); 8 waves, each owns a 16-row Q tile; streams 32-key
// blocks. maskMode 0: +1 inside |i-j|<=R (additive band); 1: keys >= Treal
// get -1e30 (padded tag table).

__global__ __launch_bounds__(256) void flash_attn_k(
    const unsigned short* __restrict__ Qb, const unsigned short* __restrict__ Kb,
    const unsigned short* __restrict__ Vb, unsigned short* __restrict__ CTX,
    int Skv, int kvBatchRows, int maskMode, int Treal) {
  __shared__ unsigned short sK[32 * 72];      // [key][d], pad 72
  __shared__ unsigned short sV[64 * 40];      // [d][key], pad 40
  __shared__ unsigned short sP[8 * 16 * 40];  // per-wave P bounce

  const int tid = threadIdx.x, lane = tid & 31, wid = tid >> 5;
  const int b = blockIdx.z, h = blockIdx.y;
  const int q0 = blockIdx.x * 128 + wid * 16;
  const int mrow = lane & 15, hi = (lane >> 4) & 1;
  const int kA = hi ? 8 : 0;

  const unsigned short* qp = Qb + (size_t)(b * Sseq + q0 + mrow) * Dm + h * DHd;
  v16bf aq0 = load_frag(qp + kA, qp + kA + 16);
  v16bf aq1 = load_frag(qp + 32 + kA, qp + 32 + kA + 16);

  v8f z = {0.f, 0.f, 0.f, 0.f, 0.f, 0.f, 0.f, 0.f};
  v8f o[4] = {z, z, z, z};
  float rm[8], rl[8];
#pragma unroll
  for (int r = 0; r < 8; ++r) { rm[r] = -3.0e38f; rl[r] = 0.0f; }

  unsigned short* myP = &sP[wid * (16 * 40)];
  const int skey = tid >> 3, sslot = tid & 7;

  for (int jb = 0; jb < Skv; jb += 32) {
    __syncthreads();
    size_t kvoff = (size_t)(kvBatchRows * b + jb + skey) * Dm + h * DHd + sslot * 8;
    cp16(Kb + kvoff, &sK[skey * 72 + sslot * 8]);  // async K [key][d]
    union { uint4 u; unsigned short us[8]; } vv;   // V transposed [d][key]
    vv.u = *(const uint4*)(Vb + kvoff);
#pragma unroll
    for (int j = 0; j < 8; ++j) sV[(sslot * 8 + j) * 40 + skey] = vv.us[j];
    async_wait0();
    __syncthreads();

    // scores: two 16x16 tiles, K-dim = DH = 64 (2 wmma each)
    float st[2][8];
#pragma unroll
    for (int kt = 0; kt < 2; ++kt) {
      int keyn = kt * 16 + mrow;
      v16bf bk0 = load_frag(&sK[keyn * 72 + hi * 8], &sK[keyn * 72 + hi * 8 + 16]);
      v16bf bk1 = load_frag(&sK[keyn * 72 + 32 + hi * 8],
                            &sK[keyn * 72 + 32 + hi * 8 + 16]);
      v8f s = z;
      s = wmma_bf16(aq0, bk0, s);
      s = wmma_bf16(aq1, bk1, s);
      int jkey = jb + keyn;
#pragma unroll
      for (int r = 0; r < 8; ++r) {
        float sc = s[r] * 0.125f; // 1/sqrt(64)
        if (maskMode == 0) {
          int dd = (q0 + r + hi * 8) - jkey;
          if (dd < 0) dd = -dd;
          sc += (dd <= Rwin) ? 1.0f : 0.0f;
        } else if (jkey >= Treal) {
          sc = -1.0e30f;
        }
        st[kt][r] = sc;
      }
    }

    // online softmax per row (row lives in one 16-lane half)
#pragma unroll
    for (int r = 0; r < 8; ++r) {
      float mx = fmaxf(st[0][r], st[1][r]);
      for (int d = 1; d < 16; d <<= 1) mx = fmaxf(mx, __shfl_xor(mx, d, 32));
      float newm = fmaxf(rm[r], mx);
      float corr = __expf(rm[r] - newm);
      float p0 = __expf(st[0][r] - newm);
      float p1 = __expf(st[1][r] - newm);
      float ps = p0 + p1;
      for (int d = 1; d < 16; d <<= 1) ps += __shfl_xor(ps, d, 32);
      rl[r] = rl[r] * corr + ps;
      rm[r] = newm;
#pragma unroll
      for (int dt = 0; dt < 4; ++dt) o[dt][r] *= corr;
      myP[(r + hi * 8) * 40 + mrow] = f2bf(p0);
      myP[(r + hi * 8) * 40 + 16 + mrow] = f2bf(p1);
    }

    // P (A-frag) x V (B-frag)
    v16bf pf = load_frag(&myP[mrow * 40 + kA], &myP[mrow * 40 + kA + 16]);
#pragma unroll
    for (int dt = 0; dt < 4; ++dt) {
      int d = dt * 16 + mrow;
      v16bf bv = load_frag(&sV[d * 40 + hi * 8], &sV[d * 40 + hi * 8 + 16]);
      o[dt] = wmma_bf16(pf, bv, o[dt]);
    }
  }

#pragma unroll
  for (int dt = 0; dt < 4; ++dt) {
#pragma unroll
    for (int r = 0; r < 8; ++r) {
      size_t grow = (size_t)(b * Sseq + q0 + r + hi * 8);
      CTX[grow * Dm + h * DHd + dt * 16 + mrow] = f2bf(o[dt][r] / rl[r]);
    }
  }
}

// ---------------- LayerNorm (row per block, N = 768) ----------------

__global__ __launch_bounds__(256) void layernorm_k(
    const float* __restrict__ in, const float* __restrict__ g,
    const float* __restrict__ beta, float* __restrict__ outF,
    unsigned short* __restrict__ outB) {
  const int N = Dm;
  int row = blockIdx.x, tid = threadIdx.x;
  const float* x = in + (size_t)row * N;
  float v0 = x[tid], v1 = x[tid + 256], v2 = x[tid + 512];
  __shared__ float r1[256], r2[256];
  r1[tid] = v0 + v1 + v2;
  r2[tid] = v0 * v0 + v1 * v1 + v2 * v2;
  __syncthreads();
  for (int s = 128; s > 0; s >>= 1) {
    if (tid < s) { r1[tid] += r1[tid + s]; r2[tid] += r2[tid + s]; }
    __syncthreads();
  }
  float mean = r1[0] * (1.0f / N);
  float var = r2[0] * (1.0f / N) - mean * mean;
  float rstd = rsqrtf(fmaxf(var, 0.0f) + 1e-12f);
  float vals[3] = {v0, v1, v2};
#pragma unroll
  for (int j = 0; j < 3; ++j) {
    int c = tid + j * 256;
    float y = (vals[j] - mean) * rstd * g[c] + beta[c];
    if (outF) outF[(size_t)row * N + c] = y;
    if (outB) outB[(size_t)row * N + c] = f2bf(y);
  }
}

// ---------------- host ----------------

static inline int cdiv_h(long a, long b) { return (int)((a + b - 1) / b); }

extern "C" void kernel_launch(void* const* d_in, const int* in_sizes, int n_in,
                              void* d_out, int out_size, void* d_ws,
                              size_t ws_size, hipStream_t stream) {
  (void)in_sizes; (void)n_in; (void)out_size; (void)ws_size;

  const float* hidden  = (const float*)d_in[0];
  const float* tag_emb = (const float*)d_in[1];
  const float* sa_wq = (const float*)d_in[2];  const float* sa_bq = (const float*)d_in[3];
  const float* sa_wk = (const float*)d_in[4];  const float* sa_bk = (const float*)d_in[5];
  const float* sa_wv = (const float*)d_in[6];  const float* sa_bv = (const float*)d_in[7];
  const float* sa_wo = (const float*)d_in[8];  const float* sa_bo = (const float*)d_in[9];
  const float* sa_ln_g = (const float*)d_in[10]; const float* sa_ln_b = (const float*)d_in[11];
  const float* ca_wq = (const float*)d_in[12]; const float* ca_bq = (const float*)d_in[13];
  const float* ca_wk = (const float*)d_in[14]; const float* ca_bk = (const float*)d_in[15];
  const float* ca_wv = (const float*)d_in[16]; const float* ca_bv = (const float*)d_in[17];
  const float* ca_wo = (const float*)d_in[18]; const float* ca_bo = (const float*)d_in[19];
  const float* ca_ln_g = (const float*)d_in[20]; const float* ca_ln_b = (const float*)d_in[21];
  const float* ff_w1 = (const float*)d_in[22]; const float* ff_b1 = (const float*)d_in[23];
  const float* ff_w2 = (const float*)d_in[24]; const float* ff_b2 = (const float*)d_in[25];
  const float* ff_ln_g = (const float*)d_in[26]; const float* ff_ln_b = (const float*)d_in[27];

  char* ws = (char*)d_ws;
  size_t off = 0;
  auto alloc = [&](size_t bytes) {
    char* p = ws + off;
    off = (off + bytes + 255) & ~(size_t)255;
    return p;
  };

  const size_t MD = (size_t)Mrows * Dm;
  unsigned short* Xbf    = (unsigned short*)alloc(MD * 2);
  unsigned short* WtSAQ  = (unsigned short*)alloc((size_t)Dm * Dm * 2);
  unsigned short* WtSAK  = (unsigned short*)alloc((size_t)Dm * Dm * 2);
  unsigned short* WtSAV  = (unsigned short*)alloc((size_t)Dm * Dm * 2);
  unsigned short* WtSAO  = (unsigned short*)alloc((size_t)Dm * Dm * 2);
  unsigned short* WtCAQ  = (unsigned short*)alloc((size_t)Dm * Dm * 2);
  unsigned short* WtCAK  = (unsigned short*)alloc((size_t)Dm * Dm * 2);
  unsigned short* WtCAV  = (unsigned short*)alloc((size_t)Dm * Dm * 2);
  unsigned short* WtCAO  = (unsigned short*)alloc((size_t)Dm * Dm * 2);
  unsigned short* ff1t   = (unsigned short*)alloc((size_t)Dm * Dff * 2);
  unsigned short* ff2t   = (unsigned short*)alloc((size_t)Dff * Dm * 2);
  unsigned short* tagbf  = (unsigned short*)alloc((size_t)64 * Dm * 2);
  unsigned short* Qbf    = (unsigned short*)alloc(MD * 2);
  unsigned short* Kbf    = (unsigned short*)alloc(MD * 2);
  unsigned short* Vbf    = (unsigned short*)alloc(MD * 2);
  unsigned short* ctxbf  = (unsigned short*)alloc(MD * 2);
  unsigned short* K2bf   = (unsigned short*)alloc((size_t)64 * Dm * 2);
  unsigned short* V2bf   = (unsigned short*)alloc((size_t)64 * Dm * 2);
  float*          PRE    = (float*)alloc(MD * 4);
  float*          YF     = (float*)alloc(MD * 4);
  unsigned short* YBF    = (unsigned short*)alloc(MD * 2);
  unsigned short* interbf= (unsigned short*)alloc((size_t)Mrows * Dff * 2);

  // ---- conversions ----
  conv_bf16_k<<<cdiv_h((long)MD, 256), 256, 0, stream>>>(hidden, Xbf, (long)MD);
  long wdd = (long)Dm * Dm;
  transpose_conv_k<<<cdiv_h(wdd, 256), 256, 0, stream>>>(sa_wq, WtSAQ, Dm, Dm);
  transpose_conv_k<<<cdiv_h(wdd, 256), 256, 0, stream>>>(sa_wk, WtSAK, Dm, Dm);
  transpose_conv_k<<<cdiv_h(wdd, 256), 256, 0, stream>>>(sa_wv, WtSAV, Dm, Dm);
  transpose_conv_k<<<cdiv_h(wdd, 256), 256, 0, stream>>>(sa_wo, WtSAO, Dm, Dm);
  transpose_conv_k<<<cdiv_h(wdd, 256), 256, 0, stream>>>(ca_wq, WtCAQ, Dm, Dm);
  transpose_conv_k<<<cdiv_h(wdd, 256), 256, 0, stream>>>(ca_wk, WtCAK, Dm, Dm);
  transpose_conv_k<<<cdiv_h(wdd, 256), 256, 0, stream>>>(ca_wv, WtCAV, Dm, Dm);
  transpose_conv_k<<<cdiv_h(wdd, 256), 256, 0, stream>>>(ca_wo, WtCAO, Dm, Dm);
  transpose_conv_k<<<cdiv_h((long)Dm * Dff, 256), 256, 0, stream>>>(ff_w1, ff1t, Dm, Dff);
  transpose_conv_k<<<cdiv_h((long)Dff * Dm, 256), 256, 0, stream>>>(ff_w2, ff2t, Dff, Dm);
  tags_pad_k<<<cdiv_h(64L * Dm, 256), 256, 0, stream>>>(tag_emb, tagbf);

  dim3 gD(cdiv_h(Mrows, 128), Dm / 64);
  dim3 gT(1, Dm / 64);
  dim3 gF1(cdiv_h(Mrows, 128), Dff / 64);
  dim3 gAtt(Sseq / 128, Hh, Bsz);

  // ---- self attention ----
  gemm_bf16_k<false><<<gD, 256, 0, stream>>>(Xbf, WtSAQ, sa_bq, nullptr, nullptr, Qbf, Mrows, Dm, Dm, 0);
  gemm_bf16_k<false><<<gD, 256, 0, stream>>>(Xbf, WtSAK, sa_bk, nullptr, nullptr, Kbf, Mrows, Dm, Dm, 0);
  gemm_bf16_k<false><<<gD, 256, 0, stream>>>(Xbf, WtSAV, sa_bv, nullptr, nullptr, Vbf, Mrows, Dm, Dm, 0);
  flash_attn_k<<<gAtt, 256, 0, stream>>>(Qbf, Kbf, Vbf, ctxbf, Sseq, Sseq, 0, Sseq);
  gemm_bf16_k<false><<<gD, 256, 0, stream>>>(ctxbf, WtSAO, sa_bo, hidden, PRE, nullptr, Mrows, Dm, Dm, 0);
  layernorm_k<<<Mrows, 256, 0, stream>>>(PRE, sa_ln_g, sa_ln_b, YF, YBF);

  // ---- cross attention (tags padded 50 -> 64; pad keys masked -inf) ----
  gemm_bf16_k<false><<<gD, 256, 0, stream>>>(YBF, WtCAQ, ca_bq, nullptr, nullptr, Qbf, Mrows, Dm, Dm, 0);
  gemm_bf16_k<true><<<gT, 256, 0, stream>>>(tagbf, WtCAK, ca_bk, nullptr, nullptr, K2bf, 64, Dm, Dm, 0);
  gemm_bf16_k<true><<<gT, 256, 0, stream>>>(tagbf, WtCAV, ca_bv, nullptr, nullptr, V2bf, 64, Dm, Dm, 0);
  flash_attn_k<<<gAtt, 256, 0, stream>>>(Qbf, K2bf, V2bf, ctxbf, 64, 0, 1, Ttag);
  gemm_bf16_k<false><<<gD, 256, 0, stream>>>(ctxbf, WtCAO, ca_bo, YF, PRE, nullptr, Mrows, Dm, Dm, 0);
  layernorm_k<<<Mrows, 256, 0, stream>>>(PRE, ca_ln_g, ca_ln_b, YF, YBF);

  // ---- FFN ----
  gemm_bf16_k<false><<<gF1, 256, 0, stream>>>(YBF, ff1t, ff_b1, nullptr, nullptr, interbf, Mrows, Dff, Dm, 1);
  gemm_bf16_k<false><<<gD, 256, 0, stream>>>(interbf, ff2t, ff_b2, YF, PRE, nullptr, Mrows, Dm, Dff, 0);
  layernorm_k<<<Mrows, 256, 0, stream>>>(PRE, ff_ln_g, ff_ln_b, (float*)d_out, nullptr);
}
